// Head_20280835571855
// MI455X (gfx1250) — matrix operand order repaired and checked by reading, not compiled, not run on hardware
//
#include <hip/hip_runtime.h>
#include <hip/hip_bf16.h>
#include <math.h>

#define B_ 8
#define T_ 2048
#define C_ 1024
#define H_ 128

typedef __attribute__((ext_vector_type(16))) __bf16 v16bf;
typedef __attribute__((ext_vector_type(8)))  __bf16 v8bf;
typedef __attribute__((ext_vector_type(4)))  __bf16 v4bf;
typedef __attribute__((ext_vector_type(8)))  float  v8f;
typedef __attribute__((ext_vector_type(4)))  int    v4i;

typedef __attribute__((address_space(1))) v4i gv4i_t;   // global int4
typedef __attribute__((address_space(3))) v4i lv4i_t;   // LDS int4

static __device__ inline v16bf cat8(v8bf lo, v8bf hi) {
  return __builtin_shufflevector(lo, hi, 0,1,2,3,4,5,6,7,8,9,10,11,12,13,14,15);
}
static __device__ inline v8f wmma_bf16(v16bf a, v16bf b, v8f c) {
  return __builtin_amdgcn_wmma_f32_16x16x32_bf16(false, a, false, b, (short)0, c, false, false);
}
// generic->AS(1): full 64-bit address; generic->AS(3): flat LDS addr low 32 bits
// ARE the LDS byte address (ISA 10.2 aperture rules).
static __device__ inline gv4i_t* as_global(const void* p) {
  return (gv4i_t*)(unsigned long long)(uintptr_t)p;
}
static __device__ inline lv4i_t* as_local(const void* p) {
  return (lv4i_t*)(unsigned int)(uintptr_t)p;
}

// ---------------------------------------------------------------------------
// Fused projection GEMMs: {q,k,v}[m][n] = sum_k x[m][k]*W{q,k,v}[k][n]
// M=B*T, K=C, N=H. 128x128 tile per 256-thread WG, K-step 32.
// grid.y selects which of the three projections; all run concurrently.
// ---------------------------------------------------------------------------
#define AS_STRIDE 40   // halves; 80B row stride -> 16B aligned frag reads
#define WT_STRIDE 40

__global__ __launch_bounds__(256) void proj_kernel(const float* __restrict__ x,
                                                   const float* __restrict__ Wq,
                                                   const float* __restrict__ Wk,
                                                   const float* __restrict__ Wv,
                                                   __bf16* __restrict__ qo,
                                                   __bf16* __restrict__ ko,
                                                   __bf16* __restrict__ vo,
                                                   float qscale) {
  __shared__ __bf16 As[128 * AS_STRIDE];   // [row][k]  (row-major, A side)
  __shared__ __bf16 Wt[128 * WT_STRIDE];   // [n][k]    (transposed, B side)
  const int t    = threadIdx.x;
  const int w    = t >> 5;
  const int lane = t & 31;
  const int mr   = lane & 15;
  const int h2   = lane >> 4;
  const int m0   = blockIdx.x * 128;

  const float* W = (blockIdx.y == 0) ? Wq : (blockIdx.y == 1) ? Wk : Wv;
  __bf16*    out = (blockIdx.y == 0) ? qo : (blockIdx.y == 1) ? ko : vo;
  const float sc = (blockIdx.y == 0) ? qscale : 1.0f;

  v8f acc[8];
  const v8f vzero = {0.f,0.f,0.f,0.f,0.f,0.f,0.f,0.f};
#pragma unroll
  for (int i = 0; i < 8; ++i) acc[i] = vzero;

  for (int kk0 = 0; kk0 < C_; kk0 += 32) {
    // x tile: 128 rows x 32 cols (float4 coalesced) -> packed b64 LDS stores
#pragma unroll
    for (int it = 0; it < 4; ++it) {
      int idx = it * 256 + t;          // 1024 float4 vectors
      int r = idx >> 3, cv = idx & 7;
      const float4 f = *(const float4*)&x[(size_t)(m0 + r) * C_ + kk0 + cv * 4];
      v4bf p;
      p[0] = (__bf16)f.x; p[1] = (__bf16)f.y; p[2] = (__bf16)f.z; p[3] = (__bf16)f.w;
      *(v4bf*)&As[r * AS_STRIDE + cv * 4] = p;
    }
    // W tile: 32 rows x 128 cols, store TRANSPOSED (Wt[n][k])
#pragma unroll
    for (int it = 0; it < 4; ++it) {
      int idx = it * 256 + t;
      int r = idx >> 5, cv = idx & 31;
      const float4 f = *(const float4*)&W[(size_t)(kk0 + r) * H_ + cv * 4];
      Wt[(cv * 4 + 0) * WT_STRIDE + r] = (__bf16)f.x;
      Wt[(cv * 4 + 1) * WT_STRIDE + r] = (__bf16)f.y;
      Wt[(cv * 4 + 2) * WT_STRIDE + r] = (__bf16)f.z;
      Wt[(cv * 4 + 3) * WT_STRIDE + r] = (__bf16)f.w;
    }
    __syncthreads();

    const __bf16* ap = &As[(w * 16 + mr) * AS_STRIDE + h2 * 8];
    v16bf a = cat8(*(const v8bf*)ap, *(const v8bf*)(ap + 16));
#pragma unroll
    for (int nt = 0; nt < 8; ++nt) {
      const __bf16* bp = &Wt[(nt * 16 + mr) * WT_STRIDE + h2 * 8];
      v16bf b = cat8(*(const v8bf*)bp, *(const v8bf*)(bp + 16));
      acc[nt] = wmma_bf16(a, b, acc[nt]);
    }
    __syncthreads();
  }

  // C/D layout: lane holds (row = v + 8*h2, col = nt*16 + mr)
#pragma unroll
  for (int nt = 0; nt < 8; ++nt)
#pragma unroll
    for (int vv = 0; vv < 8; ++vv)
      out[(size_t)(m0 + w * 16 + vv + 8 * h2) * H_ + nt * 16 + mr] =
          (__bf16)(acc[nt][vv] * sc);
}

// ---------------------------------------------------------------------------
// Flash attention: out = softmax(q k^T / sqrt(H)) v   (scale folded into q)
// 64 query rows per 128-thread WG (4 waves x 16 rows), 64-key blocks.
// K block staged via GLOBAL_LOAD_ASYNC_TO_LDS_B128 when available.
// ---------------------------------------------------------------------------
#define KB 64
#define KS_STRIDE 136   // 272B rows -> 16B aligned frag reads
#define VT_STRIDE 72    // 144B rows
#define PS_STRIDE 72

#if __has_builtin(__builtin_amdgcn_global_load_async_to_lds_b128)
#define HAVE_ASYNC_LDS 1
#else
#define HAVE_ASYNC_LDS 0
#endif

__global__ __launch_bounds__(128) void attn_kernel(const __bf16* __restrict__ q,
                                                   const __bf16* __restrict__ k,
                                                   const __bf16* __restrict__ v,
                                                   float* __restrict__ out) {
  __shared__ __bf16 Ks[KB * KS_STRIDE];     // [key][h]
  __shared__ __bf16 Vt[H_ * VT_STRIDE];     // [h][key]  (transposed)
  __shared__ __bf16 Ps[4 * 16 * PS_STRIDE]; // per-wave P scratch [qrow][key]

  const int t    = threadIdx.x;
  const int w    = t >> 5;
  const int lane = t & 31;
  const int mr   = lane & 15;
  const int h2   = lane >> 4;
  const int b    = blockIdx.y;
  const int q0   = blockIdx.x * 64 + w * 16;

  // Q fragments: 4 chunks of K=32 across H=128
  const __bf16* qbase = q + ((size_t)b * T_ + q0) * H_;
  v16bf qf[4];
#pragma unroll
  for (int h = 0; h < 4; ++h) {
    const __bf16* qp = qbase + (size_t)mr * H_ + h * 32 + h2 * 8;
    qf[h] = cat8(*(const v8bf*)qp, *(const v8bf*)(qp + 16));
  }

  const v8f vzero = {0.f,0.f,0.f,0.f,0.f,0.f,0.f,0.f};
  float m_run[8], l_run[8];
  v8f o[8];
#pragma unroll
  for (int i = 0; i < 8; ++i) { m_run[i] = -3.0e38f; l_run[i] = 0.f; o[i] = vzero; }

  const __bf16* kbase = k + (size_t)b * T_ * H_;
  const __bf16* vbase = v + (size_t)b * T_ * H_;
  __bf16* ps = &Ps[w * 16 * PS_STRIDE];

  for (int kb0 = 0; kb0 < T_; kb0 += KB) {
    __syncthreads();
    // --- stage K (row-major): async DMA global->LDS when available ---------
#pragma unroll
    for (int it = 0; it < 8; ++it) {
      int idx = it * 128 + t;          // 1024 16-byte vectors
      int r = idx >> 4, cv = idx & 15;
      const __bf16* gp = kbase + (size_t)(kb0 + r) * H_ + cv * 8;
      __bf16* lp = &Ks[r * KS_STRIDE + cv * 8];
#if HAVE_ASYNC_LDS
      __builtin_amdgcn_global_load_async_to_lds_b128(as_global(gp), as_local(lp), 0, 0);
#else
      *(v8bf*)lp = *(const v8bf*)gp;
#endif
    }
    // --- stage V transposed (overlaps with in-flight async K copies) -------
#pragma unroll
    for (int it = 0; it < 8; ++it) {
      int idx = it * 128 + t;
      int r = idx >> 4, cv = idx & 15;
      v8bf vvec = *(const v8bf*)(vbase + (size_t)(kb0 + r) * H_ + cv * 8);
#pragma unroll
      for (int i = 0; i < 8; ++i) Vt[(cv * 8 + i) * VT_STRIDE + r] = vvec[i];
    }
#if HAVE_ASYNC_LDS
    asm volatile("s_wait_asynccnt 0x0" ::: "memory");
#endif
    __syncthreads();

    // S = q k^T for 4 key subtiles (16 keys each)
    v8f s[4];
#pragma unroll
    for (int st = 0; st < 4; ++st) {
      v8f acc = vzero;
#pragma unroll
      for (int h = 0; h < 4; ++h) {
        const __bf16* bp = &Ks[(st * 16 + mr) * KS_STRIDE + h * 32 + h2 * 8];
        v16bf bf = cat8(*(const v8bf*)bp, *(const v8bf*)(bp + 16));
        acc = wmma_bf16(qf[h], bf, acc);
      }
      s[st] = acc;
    }

    // online softmax (row stats reduced across the 16-lane half via shfl_xor)
    float factor[8];
#pragma unroll
    for (int vv = 0; vv < 8; ++vv) {
      float mx = fmaxf(fmaxf(s[0][vv], s[1][vv]), fmaxf(s[2][vv], s[3][vv]));
#pragma unroll
      for (int off = 1; off < 16; off <<= 1) mx = fmaxf(mx, __shfl_xor(mx, off, 32));
      float m_new = fmaxf(m_run[vv], mx);
      factor[vv] = __expf(m_run[vv] - m_new);
      float rs = 0.f;
#pragma unroll
      for (int st = 0; st < 4; ++st) {
        float p = __expf(s[st][vv] - m_new);
        s[st][vv] = p;
        rs += p;
      }
#pragma unroll
      for (int off = 1; off < 16; off <<= 1) rs += __shfl_xor(rs, off, 32);
      l_run[vv] = l_run[vv] * factor[vv] + rs;
      m_run[vv] = m_new;
    }

    // C-layout -> A-layout for P via per-wave LDS scratch
#pragma unroll
    for (int st = 0; st < 4; ++st)
#pragma unroll
      for (int vv = 0; vv < 8; ++vv)
        ps[(vv + 8 * h2) * PS_STRIDE + st * 16 + mr] = (__bf16)s[st][vv];

#pragma unroll
    for (int nt = 0; nt < 8; ++nt)
#pragma unroll
      for (int vv = 0; vv < 8; ++vv)
        o[nt][vv] *= factor[vv];

    // intra-wave LDS visibility: DS ops are in-order, wave is lockstep
    asm volatile("s_wait_dscnt 0" ::: "memory");

    // O += P * V   (two K=32 chunks covering the 64-key block)
#pragma unroll
    for (int kc = 0; kc < 2; ++kc) {
      const __bf16* pp = &ps[mr * PS_STRIDE + kc * 32 + h2 * 8];
      v16bf pf = cat8(*(const v8bf*)pp, *(const v8bf*)(pp + 16));
#pragma unroll
      for (int nt = 0; nt < 8; ++nt) {
        const __bf16* bp = &Vt[(nt * 16 + mr) * VT_STRIDE + kc * 32 + h2 * 8];
        v16bf bf = cat8(*(const v8bf*)bp, *(const v8bf*)(bp + 16));
        o[nt] = wmma_bf16(pf, bf, o[nt]);
      }
    }
  }

  float* ob = out + ((size_t)b * T_ + q0) * H_;
#pragma unroll
  for (int nt = 0; nt < 8; ++nt)
#pragma unroll
    for (int vv = 0; vv < 8; ++vv)
      ob[(size_t)(vv + 8 * h2) * H_ + nt * 16 + mr] = o[nt][vv] / l_run[vv];
}

// ---------------------------------------------------------------------------
extern "C" void kernel_launch(void* const* d_in, const int* in_sizes, int n_in,
                              void* d_out, int out_size, void* d_ws, size_t ws_size,
                              hipStream_t stream) {
  const float* x  = (const float*)d_in[0];
  const float* Wq = (const float*)d_in[1];
  const float* Wk = (const float*)d_in[2];
  const float* Wv = (const float*)d_in[3];
  float* out = (float*)d_out;

  const size_t qkv_elems = (size_t)B_ * T_ * H_;   // 2M bf16 = 4MB each
  __bf16* qb = (__bf16*)d_ws;
  __bf16* kb = qb + qkv_elems;
  __bf16* vb = kb + qkv_elems;

  const float scale = 1.0f / sqrtf((float)H_);
  const int mblocks = (B_ * T_) / 128;             // 128

  proj_kernel<<<dim3(mblocks, 3), 256, 0, stream>>>(x, Wq, Wk, Wv, qb, kb, vb, scale);
  attn_kernel<<<dim3(T_ / 64, B_), 128, 0, stream>>>(qb, kb, vb, out);
}
